// CausalSelfAttention_64493228917129
// MI455X (gfx1250) — matrix operand order repaired and checked by reading, not compiled, not run on hardware
//
#include <hip/hip_runtime.h>
#include <hip/hip_bf16.h>
#include <math.h>

typedef __attribute__((ext_vector_type(16))) _Float16     v16h;
typedef __attribute__((ext_vector_type(8)))  _Float16     v8h;
typedef __attribute__((ext_vector_type(8)))  float        v8f;
typedef __attribute__((ext_vector_type(4)))  unsigned int uint4v;
typedef __attribute__((ext_vector_type(8)))  int          int8v;
typedef __attribute__((ext_vector_type(4)))  int          int4v;

#define BB   4
#define TT   2048
#define DDIM 1024
#define HH   16
#define HDD  64

// Compiler-level memory barrier: the TDM writes LDS behind the compiler's
// back, so force it to treat memory as modified (emits no instructions).
#define TDM_MEM_CLOBBER() asm volatile("" ::: "memory")

// ---------------------------------------------------------------------------
// WMMA helper (CDNA5 16x16x32 f16 -> f32)
// ---------------------------------------------------------------------------
__device__ __forceinline__ v8f wmma_f16(v16h a, v16h b, v8f c) {
    return __builtin_amdgcn_wmma_f32_16x16x32_f16(
        /*neg_a=*/false, a, /*neg_b=*/false, b,
        /*c_mod=*/(short)0, c, /*reuse_a=*/false, /*reuse_b=*/false);
}

// A fragment: 16x32 f16, src row-major with leading dim ldk (elements).
// lanes 0-15: row M=lane, K={0..15}; lanes 16-31: K={16..31}.
__device__ __forceinline__ v16h load_afrag(const _Float16* __restrict__ src,
                                           int ldk, int k0, int lane) {
    int row = lane & 15;
    int kb  = (lane < 16) ? 0 : 16;
    const _Float16* p = src + (size_t)row * ldk + k0 + kb;
    union { v16h v; v8h h[2]; } u;
    u.h[0] = *(const v8h*)(p);
    u.h[1] = *(const v8h*)(p + 8);
    return u.v;
}

// B fragment: 32x16 f16. Element (k, n) must live at src[n*ldk + k]:
// each lane loads 16 contiguous K for column n = lane&15 (lanes>=16: K 16..31).
__device__ __forceinline__ v16h load_bfrag(const _Float16* __restrict__ src,
                                           int ldk, int k0, int lane) {
    int col = lane & 15;
    int kb  = (lane < 16) ? 0 : 16;
    return *(const v16h*)(src + (size_t)col * ldk + k0 + kb);
}

// LDS variant without __restrict__ (wtile is written by the TDM engine).
__device__ __forceinline__ v16h load_bfrag_lds(const _Float16* src,
                                               int ldk, int k0, int lane) {
    int col = lane & 15;
    int kb  = (lane < 16) ? 0 : 16;
    return *(const v16h*)(src + (size_t)col * ldk + k0 + kb);
}

// Generic pointer into LDS -> 32-bit LDS byte address (ISA 10.2: addr[31:0]).
// The ptrtoint also makes the LDS object's address escape, so the compiler
// cannot fold loads from it to undef even though only the TDM writes it.
__device__ __forceinline__ unsigned int lds_byte_addr(const void* p) {
    return (unsigned int)(uintptr_t)p;
}

// ---------------------------------------------------------------------------
// TDM: stage a [tile1 rows x tile0 elems] f16 tile (rows contiguous in LDS)
// from a row-major tensor with leading dim tdim0, via tensor_load_to_lds.
// D# packing per CDNA5 ISA (group0 / group1 bitfields).
// ---------------------------------------------------------------------------
__device__ __forceinline__ void tdm_load_tile_f16(const _Float16* gptr,
                                                  const _Float16* lds_dst,
                                                  unsigned int tdim0,
                                                  unsigned int tdim1,
                                                  unsigned int tile0,
                                                  unsigned int tile1) {
    unsigned long long ga = (unsigned long long)(uintptr_t)gptr;
    uint4v g0 = {};
    g0.x = 1u;                                       // count=1 (valid), user mode
    g0.y = lds_byte_addr(lds_dst);                   // lds_addr (bytes)
    g0.z = (unsigned int)ga;                         // global_addr[31:0]
    g0.w = (unsigned int)((ga >> 32) & 0x1FFFFFFu)   // global_addr[56:32]
         | (2u << 30);                               // type = 2 ("image")
    int8v g1 = {};
    g1[0] = (int)(1u << 16);                                        // data_size=1 (2B)
    g1[1] = (int)((tdim0 & 0xFFFFu) << 16);                         // tensor_dim0[15:0]
    g1[2] = (int)(((tdim0 >> 16) & 0xFFFFu) | ((tdim1 & 0xFFFFu) << 16));
    g1[3] = (int)(((tdim1 >> 16) & 0xFFFFu) | ((tile0 & 0xFFFFu) << 16));
    g1[4] = (int)(tile1 & 0xFFFFu);                                 // tile_dim1, tile_dim2=0
    g1[5] = (int)tdim0;                                             // dim0 stride [31:0]
    g1[6] = 0;                                                      // stride hi / dim1 stride
    g1[7] = 0;
    int4v gz4 = {};
    int8v gz8 = {};
    __builtin_amdgcn_tensor_load_to_lds(g0, g1, gz4, gz4, gz8, 0);
    TDM_MEM_CLOBBER();
}

// ---------------------------------------------------------------------------
// fp32 -> f16 conversion
// ---------------------------------------------------------------------------
__global__ void f32_to_f16_kernel(const float* __restrict__ src,
                                  _Float16* __restrict__ dst, int n) {
    int i = blockIdx.x * blockDim.x + threadIdx.x;
    if (i < n) dst[i] = (_Float16)src[i];
}

// ---------------------------------------------------------------------------
// WMMA GEMM with TDM-staged weights: C[m,n] = sum_k A[m,k]*W[n,k] + bias[n]
//   A: [M,K] f16 row-major.  W: [N,K] f16 row-major.
// Workgroup = 4 waves. Tile M=128 (32 rows/wave), N=64, K-step 32.
// W chunk (64 n-rows x 32 k) double-buffered in LDS via tensor_load_to_lds,
// shared by all 4 waves; A fragments stream from global (private per wave).
//   mode 0: f16 out [b,h,t,hd]   mode 2: f16 out [b,h,hd,t]   mode 3: f32 [m,n]
// ---------------------------------------------------------------------------
__global__ void gemm_tdm_kernel(const _Float16* __restrict__ A,
                                const _Float16* __restrict__ W,
                                const float*    __restrict__ bias,
                                int M, int N, int K,
                                _Float16* __restrict__ outH,
                                float*    __restrict__ outF,
                                int mode) {
    __shared__ __align__(128) _Float16 wtile[2][64 * 32];   // 2 x 4KB
    int tid  = threadIdx.x;
    int lane = tid & 31;
    int wave = tid >> 5;
    int m0 = blockIdx.x * 128 + wave * 32;
    int n0 = blockIdx.y * 64;

    // Escape the LDS base so the compiler cannot treat it as write-free.
    {
        unsigned int esc = lds_byte_addr(&wtile[0][0]);
        asm volatile("" :: "s"(esc) : "memory");
    }

    v8f acc0[4] = {}, acc1[4] = {};
    const _Float16* arow0 = A + (size_t)m0 * K;
    const _Float16* arow1 = A + (size_t)(m0 + 16) * K;

    // prologue: stage k-chunk 0 into buffer 0
    if (wave == 0) {
        tdm_load_tile_f16(W + (size_t)n0 * K, &wtile[0][0],
                          (unsigned)K, (unsigned)N, 32u, 64u);
        __builtin_amdgcn_s_wait_tensorcnt(0);
        TDM_MEM_CLOBBER();
    }
    __syncthreads();
    TDM_MEM_CLOBBER();

    int nsteps = K / 32;
    for (int s = 0; s < nsteps; ++s) {
        int cur = s & 1;
        if (wave == 0 && (s + 1) < nsteps) {
            tdm_load_tile_f16(W + (size_t)n0 * K + (size_t)(s + 1) * 32,
                              &wtile[(s + 1) & 1][0],
                              (unsigned)K, (unsigned)N, 32u, 64u);
        }
        v16h a0 = load_afrag(arow0, K, s * 32, lane);
        v16h a1 = load_afrag(arow1, K, s * 32, lane);
#pragma unroll
        for (int sub = 0; sub < 4; ++sub) {
            v16h b = load_bfrag_lds(&wtile[cur][sub * 16 * 32], 32, 0, lane);
            acc0[sub] = wmma_f16(a0, b, acc0[sub]);
            acc1[sub] = wmma_f16(a1, b, acc1[sub]);
        }
        if (wave == 0 && (s + 1) < nsteps) {
            __builtin_amdgcn_s_wait_tensorcnt(0);
            TDM_MEM_CLOBBER();
        }
        __syncthreads();
        TDM_MEM_CLOBBER();
    }

    int nc   = lane & 15;
    int radd = (lane < 16) ? 0 : 8;
#pragma unroll
    for (int rt = 0; rt < 2; ++rt) {
        int mbase = m0 + rt * 16;
#pragma unroll
        for (int sub = 0; sub < 4; ++sub) {
            int n = n0 + sub * 16 + nc;
            float bval = bias ? bias[n] : 0.0f;
#pragma unroll
            for (int r = 0; r < 8; ++r) {
                int m = mbase + r + radd;
                float val = (rt == 0 ? acc0[sub][r] : acc1[sub][r]) + bval;
                if (mode == 3) {
                    outF[(size_t)m * N + n] = val;
                } else {
                    int b  = m / TT,  t  = m % TT;
                    int h  = n / HDD, hd = n % HDD;
                    if (mode == 2)
                        outH[(((size_t)(b * HH + h)) * HDD + hd) * TT + t] = (_Float16)val;
                    else
                        outH[(((size_t)(b * HH + h)) * TT + t) * HDD + hd] = (_Float16)val;
                }
            }
        }
    }
}

// ---------------------------------------------------------------------------
// RoPE in-place on q and k, layout [b,h,t,hd] f16. Positions start at 1.
// ---------------------------------------------------------------------------
__global__ void rope_kernel(_Float16* __restrict__ qh,
                            _Float16* __restrict__ kh, int total) {
    int gid = blockIdx.x * blockDim.x + threadIdx.x;
    if (gid >= total) return;            // total = B*H*T*(HD/2)
    int j   = gid & 31;
    int bht = gid >> 5;
    int t   = bht & (TT - 1);
    size_t base = (size_t)bht * HDD;
    float ang = (float)(t + 1) * __powf(10000.0f, -(float)(2 * j) / 64.0f);
    float c = __cosf(ang), s = __sinf(ang);
    float q1 = (float)qh[base + j], q2 = (float)qh[base + j + 32];
    qh[base + j]      = (_Float16)(q1 * c - q2 * s);
    qh[base + j + 32] = (_Float16)(q2 * c + q1 * s);
    float k1 = (float)kh[base + j], k2 = (float)kh[base + j + 32];
    kh[base + j]      = (_Float16)(k1 * c - k2 * s);
    kh[base + j + 32] = (_Float16)(k2 * c + k1 * s);
}

// ---------------------------------------------------------------------------
// Flash attention: one wave per (b*H+h, 16-query tile).
//   q,k: [b,h,t,hd] f16;  vT: [b,h,hd,t] f16;  out: [b,t,h*hd] f16.
// ---------------------------------------------------------------------------
__global__ void attn_kernel(const _Float16* __restrict__ qh,
                            const _Float16* __restrict__ kh,
                            const _Float16* __restrict__ vt,
                            _Float16* __restrict__ outH) {
    __shared__ __align__(64) _Float16 pbuf[16 * 32];
    int lane = threadIdx.x & 31;
    int qb = blockIdx.x * 16;
    int bh = blockIdx.y;                       // b*H + h

    const _Float16* qptr  = qh + ((size_t)bh * TT + qb) * HDD;
    const _Float16* kbase = kh + (size_t)bh * TT * HDD;
    const _Float16* vbase = vt + (size_t)bh * HDD * TT;

    v16h aq0 = load_afrag(qptr, HDD, 0, lane);
    v16h aq1 = load_afrag(qptr, HDD, 32, lane);

    v8f o0 = {}, o1 = {}, o2 = {}, o3 = {};
    float mrow[8], lrow[8];
#pragma unroll
    for (int r = 0; r < 8; ++r) { mrow[r] = -1e30f; lrow[r] = 0.0f; }

    int nc   = lane & 15;
    int radd = (lane < 16) ? 0 : 8;
    const float scale = 0.125f;                // 1/sqrt(64)

    for (int kv = 0; kv < qb + 16; kv += 32) {
        // ---- S = Q K^T for 32 keys: 2 N-subtiles x 2 K-steps
        v8f s0 = {}, s1 = {};
        {
            const _Float16* k0p = kbase + (size_t)kv * HDD;
            const _Float16* k1p = kbase + (size_t)(kv + 16) * HDD;
            s0 = wmma_f16(aq0, load_bfrag(k0p, HDD, 0,  lane), s0);
            s0 = wmma_f16(aq1, load_bfrag(k0p, HDD, 32, lane), s0);
            s1 = wmma_f16(aq0, load_bfrag(k1p, HDD, 0,  lane), s1);
            s1 = wmma_f16(aq1, load_bfrag(k1p, HDD, 32, lane), s1);
        }
        int key0 = kv + nc;
        int key1 = kv + 16 + nc;
        // ---- online softmax (row M = r+radd lives across the 16-lane half)
#pragma unroll
        for (int r = 0; r < 8; ++r) {
            int q = qb + r + radd;
            float v0 = s0[r] * scale; if (key0 > q) v0 = -1e30f;
            float v1 = s1[r] * scale; if (key1 > q) v1 = -1e30f;
            float mx = fmaxf(v0, v1);
#pragma unroll
            for (int msk = 8; msk >= 1; msk >>= 1)
                mx = fmaxf(mx, __shfl_xor(mx, msk, 32));
            float mnew = fmaxf(mrow[r], mx);
            float fac  = __expf(mrow[r] - mnew);
            float p0 = __expf(v0 - mnew);
            float p1 = __expf(v1 - mnew);
            float rs = p0 + p1;
#pragma unroll
            for (int msk = 8; msk >= 1; msk >>= 1)
                rs += __shfl_xor(rs, msk, 32);
            lrow[r] = lrow[r] * fac + rs;
            mrow[r] = mnew;
            o0[r] *= fac; o1[r] *= fac; o2[r] *= fac; o3[r] *= fac;
            int prow = r + radd;
            pbuf[prow * 32 + nc]      = (_Float16)p0;
            pbuf[prow * 32 + 16 + nc] = (_Float16)p1;
        }
        __syncthreads();
        // ---- P (16x32) x V (32x64): 4 N-subtiles over hd
        v16h pa = load_afrag(pbuf, 32, 0, lane);
        o0 = wmma_f16(pa, load_bfrag(vbase + (size_t) 0 * TT + kv, TT, 0, lane), o0);
        o1 = wmma_f16(pa, load_bfrag(vbase + (size_t)16 * TT + kv, TT, 0, lane), o1);
        o2 = wmma_f16(pa, load_bfrag(vbase + (size_t)32 * TT + kv, TT, 0, lane), o2);
        o3 = wmma_f16(pa, load_bfrag(vbase + (size_t)48 * TT + kv, TT, 0, lane), o3);
        __syncthreads();
    }

    int b = bh / HH, h = bh % HH;
#pragma unroll
    for (int r = 0; r < 8; ++r) {
        float inv = 1.0f / lrow[r];
        int t = qb + r + radd;
        size_t rowoff = ((size_t)b * TT + t) * DDIM + (size_t)h * HDD;
        outH[rowoff +  0 + nc] = (_Float16)(o0[r] * inv);
        outH[rowoff + 16 + nc] = (_Float16)(o1[r] * inv);
        outH[rowoff + 32 + nc] = (_Float16)(o2[r] * inv);
        outH[rowoff + 48 + nc] = (_Float16)(o3[r] * inv);
    }
}

// ---------------------------------------------------------------------------
// Host-side orchestration
// ---------------------------------------------------------------------------
extern "C" void kernel_launch(void* const* d_in, const int* in_sizes, int n_in,
                              void* d_out, int out_size, void* d_ws, size_t ws_size,
                              hipStream_t stream) {
    const float* x  = (const float*)d_in[0];
    const float* Wq = (const float*)d_in[1];
    const float* bq = (const float*)d_in[2];
    const float* Wk = (const float*)d_in[3];
    const float* bk = (const float*)d_in[4];
    const float* Wv = (const float*)d_in[5];
    const float* bv = (const float*)d_in[6];
    const float* Wo = (const float*)d_in[7];
    const float* bo = (const float*)d_in[8];
    float* out = (float*)d_out;

    const int NX = BB * TT * DDIM;     // 8388608
    const int NW = DDIM * DDIM;        // 1048576

    _Float16* ws = (_Float16*)d_ws;
    _Float16* xh  = ws;            ws += NX;
    _Float16* wqh = ws;            ws += NW;
    _Float16* wkh = ws;            ws += NW;
    _Float16* wvh = ws;            ws += NW;
    _Float16* woh = ws;            ws += NW;
    _Float16* qh  = ws;            ws += NX;   // [b,h,t,hd]
    _Float16* khb = ws;            ws += NX;   // [b,h,t,hd]
    _Float16* vth = ws;            ws += NX;   // [b,h,hd,t]
    _Float16* ah  = ws;            ws += NX;   // [b,t,h*hd]

    // 1) fp32 -> f16 staging
    f32_to_f16_kernel<<<(NX + 255) / 256, 256, 0, stream>>>(x,  xh,  NX);
    f32_to_f16_kernel<<<(NW + 255) / 256, 256, 0, stream>>>(Wq, wqh, NW);
    f32_to_f16_kernel<<<(NW + 255) / 256, 256, 0, stream>>>(Wk, wkh, NW);
    f32_to_f16_kernel<<<(NW + 255) / 256, 256, 0, stream>>>(Wv, wvh, NW);
    f32_to_f16_kernel<<<(NW + 255) / 256, 256, 0, stream>>>(Wo, woh, NW);

    // 2) QKV projections (M=B*T, N=D, K=D), TDM-staged weights
    const int M = BB * TT;
    dim3 ggrid(M / 128, DDIM / 64);
    gemm_tdm_kernel<<<ggrid, 128, 0, stream>>>(xh, wqh, bq, M, DDIM, DDIM, qh,  nullptr, 0);
    gemm_tdm_kernel<<<ggrid, 128, 0, stream>>>(xh, wkh, bk, M, DDIM, DDIM, khb, nullptr, 0);
    gemm_tdm_kernel<<<ggrid, 128, 0, stream>>>(xh, wvh, bv, M, DDIM, DDIM, vth, nullptr, 2);

    // 3) RoPE on q and k
    const int nrope = BB * HH * TT * (HDD / 2);
    rope_kernel<<<(nrope + 255) / 256, 256, 0, stream>>>(qh, khb, nrope);

    // 4) Flash attention
    dim3 agrid(TT / 16, BB * HH);
    attn_kernel<<<agrid, 32, 0, stream>>>(qh, khb, vth, ah);

    // 5) Output projection -> fp32 d_out
    gemm_tdm_kernel<<<ggrid, 128, 0, stream>>>(ah, woh, bo, M, DDIM, DDIM, nullptr, out, 3);
}
